// BitEuler_46368466927958
// MI455X (gfx1250) — 8-bit matrix operand order repaired and checked by reading, compiled, not run on hardware
//
#include <hip/hip_runtime.h>
#include <cstdint>


typedef int v8i __attribute__((ext_vector_type(8)));
typedef int b128v __attribute__((vector_size(16)));

#define BM 128
#define BN 128
#define BK 64
#define LDS_STRIDE 80
#define EPSQ 1e-5f

#define AS1 __attribute__((address_space(1)))
#define AS3 __attribute__((address_space(3)))

#if __has_builtin(__builtin_amdgcn_global_load_async_to_lds_b128) && \
    __has_builtin(__builtin_amdgcn_s_wait_asynccnt)
#define BE_ASYNC 1
#else
#define BE_ASYNC 0
#endif

__global__ void be_zero(float* p, int n) {
  if ((int)threadIdx.x < n) p[threadIdx.x] = 0.0f;
}

__global__ void be_abssum(const float* __restrict__ w, size_t n,
                          float* __restrict__ slot) {
  __shared__ float red[256];
  float p = 0.0f;
  for (size_t i = (size_t)blockIdx.x * blockDim.x + threadIdx.x; i < n;
       i += (size_t)gridDim.x * blockDim.x)
    p += fabsf(w[i]);
  red[threadIdx.x] = p;
  __syncthreads();
  for (int s = 128; s > 0; s >>= 1) {
    if ((int)threadIdx.x < s) red[threadIdx.x] += red[threadIdx.x + s];
    __syncthreads();
  }
  if (threadIdx.x == 0) atomicAdd(slot, red[0]);
}

__global__ void be_wquant(const float* __restrict__ w, int8_t* __restrict__ q,
                          size_t n, const float* __restrict__ sumslot,
                          float* __restrict__ gamma_out, float invcount) {
  float gamma = fmaxf(*sumslot * invcount, EPSQ);
  if (blockIdx.x == 0 && threadIdx.x == 0) *gamma_out = gamma;
  float inv = 1.0f / gamma;
  for (size_t i = (size_t)blockIdx.x * blockDim.x + threadIdx.x; i < n;
       i += (size_t)gridDim.x * blockDim.x) {
    float t = rintf(w[i] * inv);
    t = fminf(fmaxf(t, -1.0f), 1.0f);
    q[i] = (int8_t)t;
  }
}

__global__ void be_rowquant(const float* __restrict__ src, int cols,
                            int8_t* __restrict__ q, float* __restrict__ dq) {
  __shared__ float red[256];
  const int row = blockIdx.x;
  const float* r = src + (size_t)row * cols;
  float amax = 0.0f;
  for (int c = threadIdx.x; c < cols; c += blockDim.x)
    amax = fmaxf(amax, fabsf(r[c]));
  red[threadIdx.x] = amax;
  __syncthreads();
  for (int s = 128; s > 0; s >>= 1) {
    if ((int)threadIdx.x < s)
      red[threadIdx.x] = fmaxf(red[threadIdx.x], red[threadIdx.x + s]);
    __syncthreads();
  }
  float sc = fmaxf(red[0], EPSQ);
  if (threadIdx.x == 0) dq[row] = sc * (1.0f / 127.0f);
  float k = 127.0f / sc;
  int8_t* qr = q + (size_t)row * cols;
  for (int c = threadIdx.x; c < cols; c += blockDim.x) {
    float v = rintf(r[c] * k);
    v = fminf(fmaxf(v, -128.0f), 127.0f);
    qr[c] = (int8_t)v;
  }
}

__device__ __forceinline__ void be_stage(const int8_t* __restrict__ Aq,
                                         const int8_t* __restrict__ Wq,
                                         int8_t* As, int8_t* Bs, int tid,
                                         int m0, int n0, int K, int k0) {
#pragma unroll
  for (int i = 0; i < 2; ++i) {
    int c = tid + 256 * i;
    int row = c >> 2;
    int seg = (c & 3) * 16;
    const int8_t* ga = Aq + (size_t)(m0 + row) * K + k0 + seg;
    const int8_t* gb = Wq + (size_t)(n0 + row) * K + k0 + seg;
    int8_t* la = As + row * LDS_STRIDE + seg;
    int8_t* lb = Bs + row * LDS_STRIDE + seg;
#if BE_ASYNC
    __builtin_amdgcn_global_load_async_to_lds_b128(
        (AS1 b128v*)(void*)ga, (AS3 b128v*)(void*)la, 0, 0);
    __builtin_amdgcn_global_load_async_to_lds_b128(
        (AS1 b128v*)(void*)gb, (AS3 b128v*)(void*)lb, 0, 0);
#else
    *(int4*)la = *(const int4*)ga;
    *(int4*)lb = *(const int4*)gb;
#endif
  }
}

template <int MODE>
__global__ __launch_bounds__(256, 1) void be_gemm(
    const int8_t* __restrict__ Aq, const int8_t* __restrict__ Wq,
    const float* __restrict__ dqA, const float* __restrict__ gammap,
    const float* __restrict__ bias, float* __restrict__ out,
    const float* __restrict__ xin, int M, int N, int K) {
  __shared__ int8_t As[2][BM * LDS_STRIDE];
  __shared__ int8_t Bs[2][BN * LDS_STRIDE];

  const int tid  = threadIdx.x;
  const int lane = tid & 31;
  const int wave = tid >> 5;
  const int l16  = lane & 15;
  const int hi   = lane >> 4;
  const int wm   = (wave >> 2) * 64;
  const int wn   = (wave & 3) * 32;
  const int m0   = blockIdx.y * BM;
  const int n0   = blockIdx.x * BN;

  v8i acc[4][2];
  v8i vzero = {0, 0, 0, 0, 0, 0, 0, 0};
#pragma unroll
  for (int mi = 0; mi < 4; ++mi)
#pragma unroll
    for (int ni = 0; ni < 2; ++ni) acc[mi][ni] = vzero;

  be_stage(Aq, Wq, As[0], Bs[0], tid, m0, n0, K, 0);
  int buf = 0;
  for (int k0 = 0; k0 < K; k0 += BK) {
#if BE_ASYNC
    __builtin_amdgcn_s_wait_asynccnt(0);
#endif
    __syncthreads();
    if (k0 + BK < K)
      be_stage(Aq, Wq, As[buf ^ 1], Bs[buf ^ 1], tid, m0, n0, K, k0 + BK);

    v8i a[4];
#pragma unroll
    for (int mi = 0; mi < 4; ++mi) {
      const int8_t* p = &As[buf][(wm + mi * 16 + l16) * LDS_STRIDE + hi * 16];
#pragma unroll
      for (int j = 0; j < 4; ++j) {
        uint2 u = *(const uint2*)(p + 32 * (j >> 1) + 8 * (j & 1));
        a[mi][2 * j]     = (int)u.x;
        a[mi][2 * j + 1] = (int)u.y;
      }
    }
    v8i b[2];
#pragma unroll
    for (int ni = 0; ni < 2; ++ni) {
      const int8_t* p = &Bs[buf][(wn + ni * 16 + l16) * LDS_STRIDE + hi * 16];
      uint4 u0 = *(const uint4*)(p);
      uint4 u1 = *(const uint4*)(p + 32);
      v8i t = {(int)u0.x, (int)u0.y, (int)u0.z, (int)u0.w,
               (int)u1.x, (int)u1.y, (int)u1.z, (int)u1.w};
      b[ni] = t;
    }

#pragma unroll
    for (int mi = 0; mi < 4; ++mi)
#pragma unroll
      for (int ni = 0; ni < 2; ++ni)
        acc[mi][ni] = __builtin_amdgcn_wmma_i32_16x16x64_iu8(
             true, a[mi],  true, b[ni], acc[mi][ni],
             false,  false);

    buf ^= 1;
  }

  const float gamma = *gammap;
#pragma unroll
  for (int mi = 0; mi < 4; ++mi) {
    const int mbase = m0 + wm + mi * 16 + hi * 8;
    float sc[8];
#pragma unroll
    for (int j = 0; j < 8; ++j) sc[j] = dqA[mbase + j] * gamma;
#pragma unroll
    for (int ni = 0; ni < 2; ++ni) {
      int n = n0 + wn + ni * 16 + l16;
      float bn = bias[n];
#pragma unroll
      for (int j = 0; j < 8; ++j) {
        float val = (float)acc[mi][ni][j] * sc[j] + bn;
        size_t idx = (size_t)(mbase + j) * N + n;
        if (MODE == 0) {
          out[idx] = val / (1.0f + expf(-val));
        } else {
          out[idx] = xin[idx] + val * 0.1f;
        }
      }
    }
  }
}

extern "C" void kernel_launch(void* const* d_in, const int* in_sizes, int n_in,
                              void* d_out, int out_size, void* d_ws,
                              size_t ws_size, hipStream_t stream) {
  (void)in_sizes; (void)n_in; (void)out_size; (void)ws_size;
  const float* x_in = (const float*)d_in[0];
  const float* W1   = (const float*)d_in[1];
  const float* b1   = (const float*)d_in[2];
  const float* W2   = (const float*)d_in[3];
  const float* b2   = (const float*)d_in[4];
  float* x = (float*)d_out;

  const int T = 4096, F = 4096, I = 16384, ITERS = 10;
  const size_t nW1 = (size_t)I * F, nW2 = (size_t)F * I;

  uint8_t* ws = (uint8_t*)d_ws;
  size_t off = 0;
  auto take = [&](size_t bytes) -> uint8_t* {
    uint8_t* p = ws + off;
    off += (bytes + 255) & ~(size_t)255;
    return p;
  };
  int8_t* W1q = (int8_t*)take(nW1);
  int8_t* W2q = (int8_t*)take(nW2);
  int8_t* xq  = (int8_t*)take((size_t)T * F);
  int8_t* hq  = (int8_t*)take((size_t)T * I);
  float*  h   = (float*)take((size_t)T * I * 4);
  float* dqx  = (float*)take((size_t)T * 4);
  float* dqh  = (float*)take((size_t)T * 4);
  float* scal = (float*)take(16 * 4);

  be_zero<<<1, 32, 0, stream>>>(scal, 16);
  be_abssum<<<2048, 256, 0, stream>>>(W1, nW1, &scal[0]);
  be_abssum<<<2048, 256, 0, stream>>>(W2, nW2, &scal[1]);
  be_wquant<<<4096, 256, 0, stream>>>(W1, W1q, nW1, &scal[0], &scal[2],
                                      1.0f / (float)nW1);
  be_wquant<<<4096, 256, 0, stream>>>(W2, W2q, nW2, &scal[1], &scal[3],
                                      1.0f / (float)nW2);

  (void)hipMemcpyAsync(x, x_in, (size_t)T * F * sizeof(float),
                       hipMemcpyDeviceToDevice, stream);

  dim3 blk(256);
  dim3 g1(I / BN, T / BM);
  dim3 g2(F / BN, T / BM);
  for (int it = 0; it < ITERS; ++it) {
    be_rowquant<<<T, 256, 0, stream>>>(x, F, xq, dqx);
    be_gemm<0><<<g1, blk, 0, stream>>>(xq, W1q, dqx, &scal[2], b1, h, nullptr,
                                       T, I, F);
    be_rowquant<<<T, 256, 0, stream>>>(h, I, hq, dqh);
    be_gemm<1><<<g2, blk, 0, stream>>>(hq, W2q, dqh, &scal[3], b2, x, x,
                                       T, F, I);
  }
}
